// RelHetGraph_13993003450920
// MI455X (gfx1250) — matrix operand order repaired and checked by reading, not compiled, not run on hardware
//
#include <hip/hip_runtime.h>
#include <hip/hip_bf16.h>
#include <math.h>

typedef __bf16 bf16;
typedef __attribute__((ext_vector_type(16))) __bf16 v16bf;
typedef __attribute__((ext_vector_type(8)))  float  v8f;

// ---------------------------------------------------------------- helpers
__device__ __forceinline__ bf16 f2bf(float f) {
  union { float f; unsigned u; } c; c.f = f;
  unsigned r = (c.u + 0x7FFFu + ((c.u >> 16) & 1u)) >> 16;  // RNE
  union { unsigned short s; bf16 b; } o; o.s = (unsigned short)r;
  return o.b;
}

__device__ __forceinline__ void atomicMaxF(float* addr, float val) {
  // monotone-bit trick: int max for >=0, uint min for <0 (init must be -inf)
  if (val >= 0.f) atomicMax((int*)addr, __float_as_int(val));
  else            atomicMin((unsigned int*)addr, __float_as_uint(val));
}

// ---------------------------------------------------------------- cvt f32 -> bf16
__global__ void k_cvt_bf16(const float* __restrict__ in, bf16* __restrict__ out, int n) {
  int i = blockIdx.x * blockDim.x + threadIdx.x;
  if (i < n) out[i] = f2bf(in[i]);
}

// ---------------------------------------------------------------- fill
__global__ void k_fill(float* __restrict__ p, float v, int n) {
  int i = blockIdx.x * blockDim.x + threadIdx.x;
  if (i < n) p[i] = v;
}

// out[n, j] = b1[j] (+ b2[j]) (+ b3[j])   (bias pre-accumulation target)
__global__ void k_bias_init(float* __restrict__ out, const float* __restrict__ b1,
                            const float* __restrict__ b2, const float* __restrict__ b3,
                            int Nn, int Cc) {
  int i = blockIdx.x * blockDim.x + threadIdx.x;
  if (i >= Nn * Cc) return;
  int j = i % Cc;
  float v = b1[j];
  if (b2) v += b2[j];
  if (b3) v += b3[j];
  out[i] = v;
}

// ---------------------------------------------------------------- WMMA GEMM (bf16 -> f32)
// C[M,N] = (relu?)(A[M,K] @ B[K,N] + bias)  ; A,B bf16 row-major, C f32.
// 128x64 tile / block (256 thr = 8 wave32), BK = 32.
// Each wave: 4 row-subtiles x 1 col-subtile -> 4 v_wmma per k-step, B frag reused 4x.
// A stored K-contiguous, B stored transposed (N-major) in LDS so every fragment is
// two 16-byte ds_load_b128 per lane (ISA 7.12.2 layouts), no sub-register packing.
__global__ __launch_bounds__(256)
void k_gemm_bf16(const bf16* __restrict__ A, const bf16* __restrict__ B,
                 const float* __restrict__ bias, float* __restrict__ C,
                 int M, int N, int K, int relu) {
  constexpr int LDA = 40;  // padded stride (80B = 20 banks; gcd(20,64)=4 -> spread)
  constexpr int LDB = 40;
  __shared__ bf16 sA[128 * LDA];
  __shared__ bf16 sBt[64 * LDB];   // transposed: sBt[n*LDB + k]

  const int t    = threadIdx.x;
  const int lane = t & 31;
  const int wv   = t >> 5;          // 0..7
  const int bm   = blockIdx.x * 128;
  const int bn   = blockIdx.y * 64;
  const int ctile = wv & 3;         // column subtile 0..3
  const int r0    = wv >> 2;        // row subtiles r0, r0+2, r0+4, r0+6

  v8f acc[4] = {v8f{}, v8f{}, v8f{}, v8f{}};

  const int arow = t >> 2, achunk = t & 3;   // A tile: 128 rows x (4 x 8 bf16), 2 rows/thread
  const int brow = t >> 3, bchunk = t & 7;   // B tile: 32 k-rows x (8 x 8 bf16)

  const int mloc = lane & 15;
  const int kba  = (lane < 16) ? 0 : 8;      // ISA 16-bit A 16x32 layout
  const int nloc = lane & 15;
  const int kbb  = (lane < 16) ? 0 : 8;      // B: same K layout as A (0-7,16-23 / 8-15,24-31)

  for (int kb = 0; kb < K; kb += 32) {
    // ---- stage A tile (128x32, zero-padded rows beyond M)
    {
      const uint4 z = {0u, 0u, 0u, 0u};
      int r1 = bm + arow;
      uint4 v1 = z;
      if (r1 < M) v1 = *(const uint4*)(A + (long)r1 * K + kb + achunk * 8);
      *(uint4*)&sA[arow * LDA + achunk * 8] = v1;
      int r2 = bm + arow + 64;
      uint4 v2 = z;
      if (r2 < M) v2 = *(const uint4*)(A + (long)r2 * K + kb + achunk * 8);
      *(uint4*)&sA[(arow + 64) * LDA + achunk * 8] = v2;
    }
    // ---- stage B tile (32x64) transposed -> sBt[n][k]
    {
      const bf16* bp = B + (long)(kb + brow) * N + bn + bchunk * 8;
#pragma unroll
      for (int i = 0; i < 8; ++i)
        sBt[(bchunk * 8 + i) * LDB + brow] = bp[i];
    }
    // prefetch next k-block while this one is consumed (global_prefetch_b8)
    if (kb + 32 < K) {
      __builtin_prefetch(A + (long)(bm + arow) * K + kb + 32 + achunk * 8, 0, 1);
      __builtin_prefetch(B + (long)(kb + 32 + brow) * N + bn + bchunk * 8, 0, 1);
    }
    __syncthreads();

    // ---- fragments: contiguous 16B LDS chunks, no packing
    union { v16bf v; uint4 q[2]; } fb;
    fb.q[0] = *(const uint4*)&sBt[(ctile * 16 + nloc) * LDB + kbb];
    fb.q[1] = *(const uint4*)&sBt[(ctile * 16 + nloc) * LDB + kbb + 16];
#pragma unroll
    for (int s = 0; s < 4; ++s) {
      const int row = (r0 + 2 * s) * 16 + mloc;
      union { v16bf v; uint4 q[2]; } fa;
      fa.q[0] = *(const uint4*)&sA[row * LDA + kba];
      fa.q[1] = *(const uint4*)&sA[row * LDA + 16 + kba];
      acc[s] = __builtin_amdgcn_wmma_f32_16x16x32_bf16(false, fa.v, false, fb.v,
                                                       (short)0, acc[s], false, false);
    }
    __syncthreads();
  }

  // ---- store: C layout lanes 0-15 -> M=i, lanes 16-31 -> M=8+i ; N = lane&15
  const int mhi = (lane >> 4) * 8;
  const int col = bn + ctile * 16 + nloc;
  const float bv = bias ? bias[col] : 0.f;
#pragma unroll
  for (int s = 0; s < 4; ++s) {
#pragma unroll
    for (int i = 0; i < 8; ++i) {
      int r = bm + (r0 + 2 * s) * 16 + mhi + i;
      if (r < M) {
        float v = acc[s][i] + bv;
        if (relu) v = v > 0.f ? v : 0.f;
        C[(long)r * N + col] = v;
      }
    }
  }
}

// ---------------------------------------------------------------- attention logits
// al[n,h] = sum_c h[n, h*C + c] * a[h, c]
__global__ void k_attn(const float* __restrict__ h, const float* __restrict__ a,
                       float* __restrict__ al, int Nn, int H, int Cc) {
  int i = blockIdx.x * blockDim.x + threadIdx.x;
  if (i >= Nn * H) return;
  int n = i / H, hh = i - n * H;
  const float* hp = h + (long)n * H * Cc + hh * Cc;
  const float* ap = a + hh * Cc;
  float s = 0.f;
  for (int c = 0; c < Cc; ++c) s += hp[c] * ap[c];
  al[i] = s;
}

// ---------------------------------------------------------------- edge softmax
__global__ void k_edge_max(const int* __restrict__ src, const int* __restrict__ dst,
                           const float* __restrict__ al_s, const float* __restrict__ al_d,
                           float* __restrict__ m, int E, int H) {
  int i = blockIdx.x * blockDim.x + threadIdx.x;
  if (i >= E * H) return;
  int e = i / H, hh = i - e * H;
  int d = dst[e];
  float x = al_s[src[e] * H + hh] + al_d[d * H + hh];
  x = x > 0.f ? x : 0.2f * x;  // leaky_relu
  atomicMaxF(&m[d * H + hh], x);
}

__global__ void k_edge_expsum(const int* __restrict__ src, const int* __restrict__ dst,
                              const float* __restrict__ al_s, const float* __restrict__ al_d,
                              const float* __restrict__ m, float* __restrict__ ex,
                              float* __restrict__ den, int E, int H) {
  int i = blockIdx.x * blockDim.x + threadIdx.x;
  if (i >= E * H) return;
  int e = i / H, hh = i - e * H;
  int d = dst[e];
  float x = al_s[src[e] * H + hh] + al_d[d * H + hh];
  x = x > 0.f ? x : 0.2f * x;
  float v = __expf(x - m[d * H + hh]);
  ex[i] = v;
  atomicAdd(&den[d * H + hh], v);
}

// out[dst, h*C+c] += h_src[src, h*C+c] * ex[e,h] / (den[dst,h] + 1e-16)
__global__ void k_edge_scatter(const int* __restrict__ src, const int* __restrict__ dst,
                               const float* __restrict__ hsrc, const float* __restrict__ ex,
                               const float* __restrict__ den, float* __restrict__ out,
                               int E, int H, int Cc) {
  int i = blockIdx.x * blockDim.x + threadIdx.x;
  int HC = H * Cc;
  if (i >= E * HC) return;
  int e = i / HC, j = i - e * HC;
  int hh = j / Cc;
  int s = src[e], d = dst[e];
  float alpha = ex[e * H + hh] / (den[d * H + hh] + 1e-16f);
  atomicAdd(&out[(long)d * HC + j], hsrc[(long)s * HC + j] * alpha);
}

// ---------------------------------------------------------------- host orchestration
namespace {

constexpr int Dk = 768, HD1 = 512, OUTC = 256;

struct GatBufs { float *slotA, *slotB, *al_s, *al_d, *m, *den, *ex; };

inline unsigned gsz(long n) { return (unsigned)((n + 255) / 256); }

void run_gat(hipStream_t stream,
             const bf16* xsrc, int Nsrc, const bf16* xdst, int Ndst, int Kd,
             int H, int Cc, const bf16* Wbf, const float* a_s, const float* a_d,
             const int* src, const int* dst, int E, GatBufs& B, float* out) {
  const int HC = H * Cc;
  dim3 blk(256);
  k_gemm_bf16<<<dim3((Nsrc + 127) / 128, HC / 64), blk, 0, stream>>>(
      xsrc, Wbf, nullptr, B.slotA, Nsrc, HC, Kd, 0);
  const float* hdst = B.slotA;
  if (xdst != xsrc) {
    k_gemm_bf16<<<dim3((Ndst + 127) / 128, HC / 64), blk, 0, stream>>>(
        xdst, Wbf, nullptr, B.slotB, Ndst, HC, Kd, 0);
    hdst = B.slotB;
  }
  k_attn<<<gsz((long)Nsrc * H), blk, 0, stream>>>(B.slotA, a_s, B.al_s, Nsrc, H, Cc);
  k_attn<<<gsz((long)Ndst * H), blk, 0, stream>>>(hdst,    a_d, B.al_d, Ndst, H, Cc);
  k_fill<<<gsz((long)Ndst * H), blk, 0, stream>>>(B.m,  -3.0e38f, Ndst * H);
  k_fill<<<gsz((long)Ndst * H), blk, 0, stream>>>(B.den, 0.0f,    Ndst * H);
  k_edge_max<<<gsz((long)E * H), blk, 0, stream>>>(src, dst, B.al_s, B.al_d, B.m, E, H);
  k_edge_expsum<<<gsz((long)E * H), blk, 0, stream>>>(src, dst, B.al_s, B.al_d, B.m,
                                                      B.ex, B.den, E, H);
  k_edge_scatter<<<gsz((long)E * HC), blk, 0, stream>>>(src, dst, B.slotA, B.ex, B.den,
                                                        out, E, H, Cc);
}

} // namespace

extern "C" void kernel_launch(void* const* d_in, const int* in_sizes, int n_in,
                              void* d_out, int out_size, void* d_ws, size_t ws_size,
                              hipStream_t stream) {
  (void)n_in; (void)out_size; (void)ws_size;
  const float* sent_f = (const float*)d_in[0];
  const float* word_f = (const float*)d_in[1];
  const int NS = in_sizes[0] / Dk;
  const int NW = in_sizes[1] / Dk;
  const int E_sim = in_sizes[2] / 2, E_ant = in_sizes[3] / 2;
  const int E_ws = in_sizes[4], E_sw = in_sizes[6];
  const int* sim_s = (const int*)d_in[2];          const int* sim_d = sim_s + E_sim;
  const int* ant_s = (const int*)d_in[3];          const int* ant_d = ant_s + E_ant;
  const int* ws_s  = (const int*)d_in[4];          const int* ws_d  = (const int*)d_in[5];
  const int* sw_s  = (const int*)d_in[6];          const int* sw_d  = (const int*)d_in[7];

  // params: lin_sent{W,b} lin_word{W,b} c1_{sim,ant,ws,sw}{W,a_s,a_d,b} c2_{sim,ant,ws}{...}
  const float* lin_sW = (const float*)d_in[8];   const float* lin_sb = (const float*)d_in[9];
  const float* lin_wW = (const float*)d_in[10];  const float* lin_wb = (const float*)d_in[11];
  const float* c1W[4]; const float* c1as[4]; const float* c1ad[4]; const float* c1b[4];
  for (int i = 0; i < 4; ++i) {
    c1W[i]  = (const float*)d_in[12 + 4 * i];  c1as[i] = (const float*)d_in[13 + 4 * i];
    c1ad[i] = (const float*)d_in[14 + 4 * i];  c1b[i]  = (const float*)d_in[15 + 4 * i];
  }
  const float* c2W[3]; const float* c2as[3]; const float* c2ad[3]; const float* c2b[3];
  for (int i = 0; i < 3; ++i) {
    c2W[i]  = (const float*)d_in[28 + 4 * i];  c2as[i] = (const float*)d_in[29 + 4 * i];
    c2ad[i] = (const float*)d_in[30 + 4 * i];  c2b[i]  = (const float*)d_in[31 + 4 * i];
  }
  float* out = (float*)d_out;

  // ---------------- workspace bump allocator
  char* ws = (char*)d_ws; size_t off = 0;
  auto alloc = [&](size_t bytes) -> void* {
    void* p = ws + off; off = (off + bytes + 255) & ~(size_t)255; return p;
  };
  bf16* lin_sW_bf = (bf16*)alloc((size_t)Dk * Dk * 2);
  bf16* lin_wW_bf = (bf16*)alloc((size_t)Dk * Dk * 2);
  bf16* c1W_bf[4]; for (int i = 0; i < 4; ++i) c1W_bf[i] = (bf16*)alloc((size_t)Dk * HD1 * 2);
  bf16* c2W_bf[3]; for (int i = 0; i < 3; ++i) c2W_bf[i] = (bf16*)alloc((size_t)HD1 * OUTC * 2);
  bf16* sent_bf = (bf16*)alloc((size_t)NS * Dk * 2);
  bf16* word_bf = (bf16*)alloc((size_t)NW * Dk * 2);
  float* hs = (float*)alloc((size_t)NS * Dk * 4);
  float* hw = (float*)alloc((size_t)NW * Dk * 4);
  bf16* hs_bf = (bf16*)alloc((size_t)NS * Dk * 2);
  bf16* hw_bf = (bf16*)alloc((size_t)NW * Dk * 2);
  GatBufs B;
  B.slotA = (float*)alloc((size_t)NW * HD1 * 4);
  B.slotB = (float*)alloc((size_t)NW * HD1 * 4);
  float* s1 = (float*)alloc((size_t)NS * HD1 * 4);
  float* w1 = (float*)alloc((size_t)NW * HD1 * 4);
  bf16* s1_bf = (bf16*)alloc((size_t)NS * HD1 * 2);
  bf16* w1_bf = (bf16*)alloc((size_t)NW * HD1 * 2);
  int Emax = E_sim; if (E_ant > Emax) Emax = E_ant; if (E_ws > Emax) Emax = E_ws;
  if (E_sw > Emax) Emax = E_sw;
  B.ex   = (float*)alloc((size_t)Emax * 8 * 4);
  B.al_s = (float*)alloc((size_t)NW * 8 * 4);
  B.al_d = (float*)alloc((size_t)NW * 8 * 4);
  B.m    = (float*)alloc((size_t)NW * 8 * 4);
  B.den  = (float*)alloc((size_t)NW * 8 * 4);

  dim3 blk(256);
  // ---------------- weight + input conversion to bf16
  k_cvt_bf16<<<gsz((long)Dk * Dk), blk, 0, stream>>>(lin_sW, lin_sW_bf, Dk * Dk);
  k_cvt_bf16<<<gsz((long)Dk * Dk), blk, 0, stream>>>(lin_wW, lin_wW_bf, Dk * Dk);
  for (int i = 0; i < 4; ++i)
    k_cvt_bf16<<<gsz((long)Dk * HD1), blk, 0, stream>>>(c1W[i], c1W_bf[i], Dk * HD1);
  for (int i = 0; i < 3; ++i)
    k_cvt_bf16<<<gsz((long)HD1 * OUTC), blk, 0, stream>>>(c2W[i], c2W_bf[i], HD1 * OUTC);
  k_cvt_bf16<<<gsz((long)NS * Dk), blk, 0, stream>>>(sent_f, sent_bf, NS * Dk);
  k_cvt_bf16<<<gsz((long)NW * Dk), blk, 0, stream>>>(word_f, word_bf, NW * Dk);

  // ---------------- input projections: hs = relu(x@W+b), hw likewise (WMMA)
  k_gemm_bf16<<<dim3((NS + 127) / 128, Dk / 64), blk, 0, stream>>>(
      sent_bf, lin_sW_bf, lin_sb, hs, NS, Dk, Dk, 1);
  k_gemm_bf16<<<dim3((NW + 127) / 128, Dk / 64), blk, 0, stream>>>(
      word_bf, lin_wW_bf, lin_wb, hw, NW, Dk, Dk, 1);
  k_cvt_bf16<<<gsz((long)NS * Dk), blk, 0, stream>>>(hs, hs_bf, NS * Dk);
  k_cvt_bf16<<<gsz((long)NW * Dk), blk, 0, stream>>>(hw, hw_bf, NW * Dk);

  // ---------------- conv1: s1 = gat_sim + gat_ant + gat_ws  (bias pre-summed)
  k_bias_init<<<gsz((long)NS * HD1), blk, 0, stream>>>(s1, c1b[0], c1b[1], c1b[2], NS, HD1);
  run_gat(stream, hs_bf, NS, hs_bf, NS, Dk, 8, 64, c1W_bf[0], c1as[0], c1ad[0],
          sim_s, sim_d, E_sim, B, s1);
  run_gat(stream, hs_bf, NS, hs_bf, NS, Dk, 8, 64, c1W_bf[1], c1as[1], c1ad[1],
          ant_s, ant_d, E_ant, B, s1);
  run_gat(stream, hw_bf, NW, hs_bf, NS, Dk, 8, 64, c1W_bf[2], c1as[2], c1ad[2],
          ws_s, ws_d, E_ws, B, s1);
  // w1 = gat_sw
  k_bias_init<<<gsz((long)NW * HD1), blk, 0, stream>>>(w1, c1b[3], nullptr, nullptr, NW, HD1);
  run_gat(stream, hs_bf, NS, hw_bf, NW, Dk, 8, 64, c1W_bf[3], c1as[3], c1ad[3],
          sw_s, sw_d, E_sw, B, w1);

  // ---------------- conv2: out = gat_sim + gat_ant + gat_ws on (s1, w1)
  k_cvt_bf16<<<gsz((long)NS * HD1), blk, 0, stream>>>(s1, s1_bf, NS * HD1);
  k_cvt_bf16<<<gsz((long)NW * HD1), blk, 0, stream>>>(w1, w1_bf, NW * HD1);
  k_bias_init<<<gsz((long)NS * OUTC), blk, 0, stream>>>(out, c2b[0], c2b[1], c2b[2], NS, OUTC);
  run_gat(stream, s1_bf, NS, s1_bf, NS, HD1, 1, OUTC, c2W_bf[0], c2as[0], c2ad[0],
          sim_s, sim_d, E_sim, B, out);
  run_gat(stream, s1_bf, NS, s1_bf, NS, HD1, 1, OUTC, c2W_bf[1], c2as[1], c2ad[1],
          ant_s, ant_d, E_ant, B, out);
  run_gat(stream, w1_bf, NW, s1_bf, NS, HD1, 1, OUTC, c2W_bf[2], c2as[2], c2ad[2],
          ws_s, ws_d, E_ws, B, out);
}
